// AlchemicalModel_65910568125206
// MI455X (gfx1250) — compile-verified, matrix operand order checked by reading, not run on hardware
//
#include <hip/hip_runtime.h>
#include <hip/hip_bf16.h>
#include <stdint.h>

// ---------------- model constants ----------------
#define N_ATOMS 16384
#define N_EDGES 524288
#define N_MOL   256
#define A_CH    4
#define S_SP    4
#define NMAX    10
#define NSPH    9
#define FDIM    4800      // 40*40*3
#define H1D     512
#define H2D     512
#define TM      32        // atoms per MLP workgroup (2 M-tiles)

typedef __attribute__((ext_vector_type(16))) __bf16 v16bf;
typedef __attribute__((ext_vector_type(8)))  float  v8f;

union FragBF { v16bf v; uint4 u[2]; };

__device__ __forceinline__ unsigned short f2bf(float f) {
  unsigned u = __builtin_bit_cast(unsigned, f);
  u += 0x7FFFu + ((u >> 16) & 1u);            // round-to-nearest-even
  return (unsigned short)(u >> 16);
}
__device__ __forceinline__ float bf2f(unsigned short h) {
  unsigned u = ((unsigned)h) << 16;
  return __builtin_bit_cast(float, u);
}
// 16-bf16 fragment: two contiguous 16B loads (b128 path), matches CDNA5
// 16-bit A/B VGPR layout (lanes 0-15: K k0..k0+7 & k0+16..23; lanes 16-31: +8)
__device__ __forceinline__ v16bf load_frag(const unsigned short* p) {
  FragBF f;
  f.u[0] = *reinterpret_cast<const uint4*>(p);
  f.u[1] = *reinterpret_cast<const uint4*>(p + 16);
  return f.v;
}
__device__ __forceinline__ float silu(float x) { return x / (1.0f + __expf(-x)); }

// power-spectrum element f = q*120 + p*3 + l  over c-row [40*9]
__device__ __forceinline__ float psval(const float* crow, int f) {
  int q = f / 120;
  int r = f - q * 120;
  int p = r / 3;
  int l = r - p * 3;
  const int   m0t[3]  = {0, 1, 4};
  const int   mlt[3]  = {1, 3, 5};
  const float invt[3] = {1.0f, 0.57735026918962576f, 0.44721359549995794f};
  const float* cq = crow + q * 9 + m0t[l];
  const float* cp = crow + p * 9 + m0t[l];
  float s = 0.f;
  for (int m = 0; m < mlt[l]; ++m) s += cq[m] * cp[m];
  return s * invt[l];
}

// ================= edge binning (1 atomic per edge instead of 90) =================
__global__ void k_ecount(const int* __restrict__ ei, int* __restrict__ ecnt) {
  int e = blockIdx.x * 256 + threadIdx.x;
  if (e < N_EDGES) atomicAdd(&ecnt[ei[N_EDGES + e]], 1);   // bin by recv
}
// exclusive prefix over 16384 counts, single block of 256 threads (64 each)
__global__ void __launch_bounds__(256) k_escan(const int* __restrict__ ecnt,
                                               int* __restrict__ eofs) {
  __shared__ int part[256];
  int t = threadIdx.x;
  int s = 0;
  for (int i = 0; i < 64; ++i) s += ecnt[t * 64 + i];
  part[t] = s;
  __syncthreads();
  if (t == 0) {
    int run = 0;
    for (int i = 0; i < 256; ++i) { int v = part[i]; part[i] = run; run += v; }
  }
  __syncthreads();
  int run = part[t];
  for (int i = 0; i < 64; ++i) { int v = ecnt[t * 64 + i]; eofs[t * 64 + i] = run; run += v; }
}
__global__ void k_eplace(const int* __restrict__ ei, const int* __restrict__ eofs,
                         int* __restrict__ ecur, int* __restrict__ eperm) {
  int e = blockIdx.x * 256 + threadIdx.x;
  if (e >= N_EDGES) return;
  int rcv = ei[N_EDGES + e];
  int pos = atomicAdd(&ecur[rcv], 1);
  eperm[eofs[rcv] + pos] = e;
}

// ---- per-atom edge accumulation: one wave per atom, LDS private row, NO atomics ----
__global__ void __launch_bounds__(256) k_edge2(
    const float* __restrict__ pos, const float* __restrict__ cells,
    const int* __restrict__ numbers, const int* __restrict__ ei,
    const float* __restrict__ eoff, const int* __restrict__ batch,
    const int* __restrict__ eperm, const int* __restrict__ eofs,
    const int* __restrict__ ecnt, float* __restrict__ cs /* [atom][4][10][9] */)
{
  __shared__ float cacc[8][360];
  int wave = threadIdx.x >> 5, lane = threadIdx.x & 31;
  int atom = blockIdx.x * 8 + wave;
  for (int idx = threadIdx.x; idx < 8 * 360; idx += 256) ((float*)cacc)[idx] = 0.f;
  __syncthreads();

  float px = pos[atom * 3 + 0], py = pos[atom * 3 + 1], pz = pos[atom * 3 + 2];
  int e0 = eofs[atom], ne = ecnt[atom];
  const float c1 = 0.4886025119029199f;
  const float c2 = 1.0925484305920792f;

  for (int i = 0; i < ne; ++i) {             // serial over this atom's edges
    int e   = eperm[e0 + i];                 // broadcast load (all lanes same)
    int snd = ei[e];
    const float* cell = cells + (size_t)batch[snd] * 9;
    float ox = eoff[e*3+0], oy = eoff[e*3+1], oz = eoff[e*3+2];
    float rx = px - pos[snd*3+0] + ox*cell[0] + oy*cell[3] + oz*cell[6];
    float ry = py - pos[snd*3+1] + ox*cell[1] + oy*cell[4] + oz*cell[7];
    float rz = pz - pos[snd*3+2] + ox*cell[2] + oy*cell[5] + oz*cell[8];
    float r  = sqrtf(rx*rx + ry*ry + rz*rz + 1e-12f);
    float fc = (r < 5.0f) ? 0.5f * (__cosf(0.62831853071795864f * r) + 1.0f) : 0.0f;
    if (fc == 0.0f) continue;
    float inv = 1.0f / r;
    float x = rx * inv, y = ry * inv, z = rz * inv;
    float R[NMAX];
#pragma unroll
    for (int n = 0; n < NMAX; ++n) {
      float d = r - (5.0f / 9.0f) * (float)n;
      R[n] = __expf(-2.0f * d * d) * fc;     // sig=0.5 -> 1/(2 sig^2)=2
    }
    float Y[NSPH];
    Y[0] = 0.28209479177387814f;
    Y[1] = c1 * y;  Y[2] = c1 * z;  Y[3] = c1 * x;
    Y[4] = c2 * x * y;  Y[5] = c2 * y * z;
    Y[6] = 0.31539156525252005f * (3.0f * z * z - 1.0f);
    Y[7] = c2 * x * z;
    Y[8] = 0.5462742152960396f * (x * x - y * y);
    int base = numbers[snd] * 90;
    // lanes own disjoint elements j, j+32, j+64 -> plain LDS RMW, no atomics
    for (int j = lane; j < 90; j += 32) {
      int n = j / 9, k = j - n * 9;
      cacc[wave][base + j] += R[n] * Y[k];
    }
  }
  // wave-private row: no barrier needed before writing it out
  float* dst = cs + (size_t)atom * 360;
  for (int j = lane; j < 360; j += 32) dst[j] = cacc[wave][j];
}

// ---------------- atom species binning + composition term ----------------
__global__ void k_count(const int* __restrict__ numbers, int* __restrict__ meta) {
  int i = blockIdx.x * 256 + threadIdx.x;
  if (i < N_ATOMS) atomicAdd(&meta[numbers[i]], 1);
}
__global__ void k_offsets(int* __restrict__ meta) {
  if (threadIdx.x == 0 && blockIdx.x == 0) {
    meta[4] = 0;
    meta[5] = meta[4] + meta[0];
    meta[6] = meta[5] + meta[1];
    meta[7] = meta[6] + meta[2];
  }
}
__global__ void k_place(const int* __restrict__ numbers, const int* __restrict__ batch,
                        const float* __restrict__ Wc, int* __restrict__ meta,
                        int* __restrict__ perm, float* __restrict__ out) {
  int i = blockIdx.x * 256 + threadIdx.x;
  if (i >= N_ATOMS) return;
  int s = numbers[i];
  int pos = atomicAdd(&meta[8 + s], 1);
  perm[meta[4 + s] + pos] = i;
  atomicAdd(&out[batch[i]], Wc[s]);   // comp @ Wc.T contribution
}

// ---------------- fold U into weights, transpose to [s][o][k], cast bf16 ------
__global__ void __launch_bounds__(256) k_wt1(const float* __restrict__ W1,
                                             const float* __restrict__ U,
                                             unsigned short* __restrict__ Wt1) {
  size_t idx = (size_t)blockIdx.x * 256 + threadIdx.x;        // [s][o][f]
  if (idx >= (size_t)S_SP * H1D * FDIM) return;
  int f = (int)(idx % FDIM);
  size_t t = idx / FDIM;
  int o = (int)(t % H1D);
  int s = (int)(t / H1D);
  float v = 0.f;
#pragma unroll
  for (int a = 0; a < A_CH; ++a)
    v += U[a * 4 + s] * W1[((size_t)a * FDIM + f) * H1D + o];
  Wt1[idx] = f2bf(v);
}
__global__ void __launch_bounds__(256) k_wt2(const float* __restrict__ W2,
                                             const float* __restrict__ U,
                                             unsigned short* __restrict__ Wt2) {
  size_t idx = (size_t)blockIdx.x * 256 + threadIdx.x;        // [s][o][i]
  if (idx >= (size_t)S_SP * H2D * H1D) return;
  int i = (int)(idx % H1D);
  size_t t = idx / H1D;
  int o = (int)(t % H2D);
  int s = (int)(t / H2D);
  float v = 0.f;
#pragma unroll
  for (int a = 0; a < A_CH; ++a)
    v += U[a * 4 + s] * W2[((size_t)a * H1D + i) * H2D + o];
  Wt2[idx] = f2bf(v);
}
__global__ void k_w3(const float* __restrict__ W3, const float* __restrict__ U,
                     float* __restrict__ w3g) {
  int idx = blockIdx.x * 256 + threadIdx.x;                   // [s][i]
  if (idx >= S_SP * H2D) return;
  int i = idx % H2D, s = idx / H2D;
  float v = 0.f;
#pragma unroll
  for (int a = 0; a < A_CH; ++a) v += U[a * 4 + s] * W3[a * H2D + i];
  w3g[idx] = v;
}

// ====== fused (U*c_s) -> ps -> LN -> MLP (bf16 WMMA, 32 atoms/block) -> segsum ======
__global__ void __launch_bounds__(256) k_mlp(
    const int* __restrict__ perm, const int* __restrict__ meta,
    const unsigned short* __restrict__ Wt1, const unsigned short* __restrict__ Wt2,
    const float* __restrict__ w3g, const float* __restrict__ cs,
    const float* __restrict__ U, const float* __restrict__ gamma,
    const float* __restrict__ beta, const int* __restrict__ batch,
    float* __restrict__ out)
{
  extern __shared__ char smem[];
  float*          cbuf  = (float*)smem;                           // 32*360*4 = 46080
  unsigned short* Xc    = (unsigned short*)(smem + 46080);        // 32*160*2 = 10240
  unsigned short* Hb    = (unsigned short*)(smem + 56320);        // 32*512*2 = 32768
  float*          red1  = (float*)(smem + 89088);                 // 32*8*4   = 1024
  float*          red2  = (float*)(smem + 90112);                 // 1024
  float*          meanv = (float*)(smem + 91136);                 // 128
  float*          rstdv = (float*)(smem + 91264);                 // 128 -> total 91392

  int cnt[4], offs[4];
#pragma unroll
  for (int j = 0; j < 4; ++j) { cnt[j] = meta[j]; offs[j] = meta[4 + j]; }

  int b = blockIdx.x, sp = -1, tloc = 0;
  {
    int acc = 0;
#pragma unroll
    for (int j = 0; j < 4; ++j) {
      int t = (cnt[j] + 31) >> 5;
      if (sp < 0 && b < acc + t) { sp = j; tloc = b - acc; }
      acc += t;
    }
  }
  if (sp < 0) return;                       // uniform across block
  int nvalid = cnt[sp] - tloc * TM; if (nvalid > TM) nvalid = TM;
  int rowBase = offs[sp] + tloc * TM;
  int tid = threadIdx.x, lane = tid & 31, wave = tid >> 5;
  int hi = lane >> 4, ln = lane & 15;

  // ---- c = sum_s U[a,s] * c_s  (32 atoms x 360) ----
  for (int idx = tid; idx < TM * 360; idx += 256) {
    int a = idx / 360, j = idx - a * 360;
    float v = 0.f;
    if (a < nvalid) {
      int atom = perm[rowBase + a];
      const float* cp = cs + (size_t)atom * 360;
      int ach = j / 90, nk = j - ach * 90;
      v = U[ach*4+0]*cp[nk] + U[ach*4+1]*cp[90+nk] + U[ach*4+2]*cp[180+nk] + U[ach*4+3]*cp[270+nk];
    }
    cbuf[idx] = v;
  }
  __syncthreads();

  // ---- LayerNorm statistics over F=4800 (8 partial threads per atom) ----
  {
    int a = tid & 31, sub = tid >> 5;
    const float* crow = cbuf + a * 360;
    float s1 = 0.f, s2 = 0.f;
    for (int f = sub; f < FDIM; f += 8) { float p = psval(crow, f); s1 += p; s2 += p * p; }
    red1[a * 8 + sub] = s1; red2[a * 8 + sub] = s2;
  }
  __syncthreads();
  if (tid < TM) {
    float s1 = 0.f, s2 = 0.f;
    for (int j = 0; j < 8; ++j) { s1 += red1[tid * 8 + j]; s2 += red2[tid * 8 + j]; }
    float mean = s1 * (1.0f / FDIM);
    float var  = s2 * (1.0f / FDIM) - mean * mean;
    meanv[tid] = mean;
    rstdv[tid] = rsqrtf(var + 1e-5f);
  }
  __syncthreads();

  const v8f z8 = {0.f,0.f,0.f,0.f,0.f,0.f,0.f,0.f};

  // ---- Layer 1: X(32x4800) @ W(4800x512); K streamed in chunks of 160 ----
  v8f acc1[2][4] = {{z8, z8, z8, z8}, {z8, z8, z8, z8}};
  for (int ch = 0; ch < 30; ++ch) {
    int f0 = ch * 160;
    for (int idx = tid; idx < TM * 160; idx += 256) {
      int a = idx / 160, j = idx - a * 160;
      int fg = f0 + j;
      float p  = psval(cbuf + a * 360, fg);
      float xn = (p - meanv[a]) * rstdv[a] * gamma[fg] + beta[fg];
      Xc[idx] = f2bf(xn);
    }
    __syncthreads();
#pragma unroll
    for (int kk = 0; kk < 5; ++kk) {
      int kb = kk * 32 + hi * 8;
      v16bf A0 = load_frag(&Xc[(ln)      * 160 + kb]);
      v16bf A1 = load_frag(&Xc[(16 + ln) * 160 + kb]);
      int kg = f0 + kb;
#pragma unroll
      for (int t = 0; t < 4; ++t) {
        int o = (wave * 4 + t) * 16 + ln;
        v16bf Bf = load_frag(Wt1 + ((size_t)(sp * H1D + o)) * FDIM + kg);
        acc1[0][t] = __builtin_amdgcn_wmma_f32_16x16x32_bf16(
            false, A0, false, Bf, (short)0, acc1[0][t], false, false);
        acc1[1][t] = __builtin_amdgcn_wmma_f32_16x16x32_bf16(
            false, A1, false, Bf, (short)0, acc1[1][t], false, false);
      }
    }
    __syncthreads();
  }
#pragma unroll
  for (int mt = 0; mt < 2; ++mt)
#pragma unroll
    for (int t = 0; t < 4; ++t)
#pragma unroll
      for (int r = 0; r < 8; ++r) {
        int M = mt * 16 + r + hi * 8;
        int N = (wave * 4 + t) * 16 + ln;
        Hb[M * 512 + N] = f2bf(silu(acc1[mt][t][r]));
      }
  __syncthreads();

  // ---- Layer 2: H1(32x512) @ W(512x512) ----
  v8f acc2[2][4] = {{z8, z8, z8, z8}, {z8, z8, z8, z8}};
#pragma unroll 4
  for (int kk = 0; kk < 16; ++kk) {
    int kb = kk * 32 + hi * 8;
    v16bf A0 = load_frag(&Hb[(ln)      * 512 + kb]);
    v16bf A1 = load_frag(&Hb[(16 + ln) * 512 + kb]);
#pragma unroll
    for (int t = 0; t < 4; ++t) {
      int o = (wave * 4 + t) * 16 + ln;
      v16bf Bf = load_frag(Wt2 + ((size_t)(sp * H2D + o)) * H1D + kb);
      acc2[0][t] = __builtin_amdgcn_wmma_f32_16x16x32_bf16(
          false, A0, false, Bf, (short)0, acc2[0][t], false, false);
      acc2[1][t] = __builtin_amdgcn_wmma_f32_16x16x32_bf16(
          false, A1, false, Bf, (short)0, acc2[1][t], false, false);
    }
  }
  __syncthreads();   // everyone done reading Hb
#pragma unroll
  for (int mt = 0; mt < 2; ++mt)
#pragma unroll
    for (int t = 0; t < 4; ++t)
#pragma unroll
      for (int r = 0; r < 8; ++r) {
        int M = mt * 16 + r + hi * 8;
        int N = (wave * 4 + t) * 16 + ln;
        Hb[M * 512 + N] = f2bf(silu(acc2[mt][t][r]));
      }
  __syncthreads();

  // ---- Layer 3 (512 -> 1) + molecule segment-sum ----
  {
    int a = tid & 31, sub = tid >> 5;
    float part = 0.f;
    for (int i = sub; i < 512; i += 8)
      part += bf2f(Hb[a * 512 + i]) * w3g[sp * 512 + i];
    red1[a * 8 + sub] = part;
  }
  __syncthreads();
  if (tid < TM && tid < nvalid) {
    float v = 0.f;
    for (int j = 0; j < 8; ++j) v += red1[tid * 8 + j];
    int atom = perm[rowBase + tid];
    atomicAdd(&out[batch[atom]], v * 0.0078125f);   // 1/sqrt(4)/64 * SCALE
  }
}

// ---------------- host-side orchestration ----------------
extern "C" void kernel_launch(void* const* d_in, const int* in_sizes, int n_in,
                              void* d_out, int out_size, void* d_ws, size_t ws_size,
                              hipStream_t stream) {
  const float* positions = (const float*)d_in[0];
  const float* cells     = (const float*)d_in[1];
  const int*   numbers   = (const int*)  d_in[2];
  const int*   ei        = (const int*)  d_in[3];
  const float* eoff      = (const float*)d_in[4];
  const int*   batch     = (const int*)  d_in[5];
  const float* U         = (const float*)d_in[6];
  const float* gamma     = (const float*)d_in[7];
  const float* beta      = (const float*)d_in[8];
  const float* W1        = (const float*)d_in[9];
  const float* W2        = (const float*)d_in[10];
  const float* W3        = (const float*)d_in[11];
  const float* Wc        = (const float*)d_in[12];
  float* out = (float*)d_out;

  // workspace layout (16B aligned offsets)
  char* ws = (char*)d_ws;
  float*          cs    = (float*)(ws + 0);                    // 23,592,960 B
  unsigned short* Wt1   = (unsigned short*)(ws + 23592960);    // 19,660,800 B
  unsigned short* Wt2   = (unsigned short*)(ws + 43253760);    //  2,097,152 B
  float*          w3g   = (float*)(ws + 45350912);             //      8,192 B
  int*            perm  = (int*)(ws + 45359104);               //     65,536 B
  int*            meta  = (int*)(ws + 45424640);               //         64 B
  int*            ecnt  = (int*)(ws + 45424704);               //     65,536 B
  int*            eofs  = (int*)(ws + 45490240);               //     65,536 B
  int*            ecur  = (int*)(ws + 45555776);               //     65,536 B
  int*            eperm = (int*)(ws + 45621312);               //  2,097,152 B

  hipMemsetAsync(meta, 0, 16 * sizeof(int), stream);
  hipMemsetAsync(ecnt, 0, N_ATOMS * sizeof(int), stream);
  hipMemsetAsync(ecur, 0, N_ATOMS * sizeof(int), stream);
  hipMemsetAsync(out,  0, (size_t)out_size * sizeof(float), stream);

  // edge binning by receiver, then atomic-free per-atom accumulation
  k_ecount<<<N_EDGES / 256, 256, 0, stream>>>(ei, ecnt);
  k_escan <<<1, 256, 0, stream>>>(ecnt, eofs);
  k_eplace<<<N_EDGES / 256, 256, 0, stream>>>(ei, eofs, ecur, eperm);
  k_edge2 <<<N_ATOMS / 8, 256, 0, stream>>>(positions, cells, numbers, ei, eoff,
                                            batch, eperm, eofs, ecnt, cs);

  // atom species binning + composition term
  k_count  <<<N_ATOMS / 256, 256, 0, stream>>>(numbers, meta);
  k_offsets<<<1, 32, 0, stream>>>(meta);
  k_place  <<<N_ATOMS / 256, 256, 0, stream>>>(numbers, batch, Wc, meta, perm, out);

  // weight preparation
  {
    size_t tot1 = (size_t)S_SP * H1D * FDIM;
    k_wt1<<<(unsigned)((tot1 + 255) / 256), 256, 0, stream>>>(W1, U, Wt1);
    size_t tot2 = (size_t)S_SP * H2D * H1D;
    k_wt2<<<(unsigned)((tot2 + 255) / 256), 256, 0, stream>>>(W2, U, Wt2);
    k_w3 <<<(S_SP * H2D + 255) / 256, 256, 0, stream>>>(W3, U, w3g);
  }

  // fused MLP: 32 atoms per block, dynamic LDS = 91392 B
  k_mlp<<<N_ATOMS / TM + 4, 256, 91392, stream>>>(perm, meta, Wt1, Wt2, w3g, cs,
                                                  U, gamma, beta, batch, out);
}
